// MoeLayer_36627481101232
// MI455X (gfx1250) — compile-verified, matrix operand order checked by reading, not run on hardware
//
#include <hip/hip_runtime.h>
#include <hip/hip_bf16.h>

#define D_MODEL   1024
#define HIDDEN    4096
#define N_EXPERTS 8
#define T_TOKENS  16384   // BATCH*SEQ = 4*4096

typedef __bf16 bf16_t;
typedef __attribute__((ext_vector_type(16))) __bf16 v16bf;
typedef __attribute__((ext_vector_type(8)))  __bf16 v8bf;
typedef __attribute__((ext_vector_type(8)))  float  v8f;
typedef unsigned int  u32x4 __attribute__((ext_vector_type(4)));
typedef int           i32x4 __attribute__((ext_vector_type(4)));
typedef int           i32x8 __attribute__((ext_vector_type(8)));

#define BM 128
#define BN 128
#define BK 64
#define LDS_STRIDE (BK + 8)   // 72 elems = 144B rows: 16B aligned, conflict-free frag reads
#define STAGE_A (BM * LDS_STRIDE)
#define STAGE_B (BN * LDS_STRIDE)

#define AS1 __attribute__((address_space(1)))
#define AS3 __attribute__((address_space(3)))

#if defined(__gfx1250__) && __has_builtin(__builtin_amdgcn_global_load_async_to_lds_b128) && \
    __has_builtin(__builtin_amdgcn_s_wait_asynccnt)
#define USE_ASYNC_LDS 1
#else
#define USE_ASYNC_LDS 0
#endif

#if defined(__gfx1250__) && __has_builtin(__builtin_amdgcn_tensor_load_to_lds) && \
    __has_builtin(__builtin_amdgcn_s_wait_tensorcnt)
#define USE_TDM 1
#else
#define USE_TDM 0
#endif

// ---------------------------------------------------------------- converts
__global__ void cvt_x_kernel(const float* __restrict__ in, bf16_t* __restrict__ out) {
    size_t i = ((size_t)blockIdx.x * blockDim.x + threadIdx.x) * 4;
    float4 v = *(const float4*)(in + i);
    union { bf16_t b[4]; unsigned long long u; } o;
    o.b[0] = (bf16_t)v.x; o.b[1] = (bf16_t)v.y; o.b[2] = (bf16_t)v.z; o.b[3] = (bf16_t)v.w;
    *(unsigned long long*)(out + i) = o.u;
}

// w1: [E][K=1024][N=4096] -> w1t: [E][N][K] bf16
__global__ void cvt_w1t_kernel(const float* __restrict__ w, bf16_t* __restrict__ wt) {
    size_t idx = (size_t)blockIdx.x * blockDim.x + threadIdx.x;
    int n = (int)(idx % HIDDEN);
    size_t r = idx / HIDDEN;
    int k = (int)(r % D_MODEL);
    int e = (int)(r / D_MODEL);
    wt[((size_t)e * HIDDEN + n) * D_MODEL + k] = (bf16_t)w[idx];
}

// w2: [E][K=4096][N=1024] -> w2t: [E][N][K] bf16
__global__ void cvt_w2t_kernel(const float* __restrict__ w, bf16_t* __restrict__ wt) {
    size_t idx = (size_t)blockIdx.x * blockDim.x + threadIdx.x;
    int n = (int)(idx % D_MODEL);
    size_t r = idx / D_MODEL;
    int k = (int)(r % HIDDEN);
    int e = (int)(r / HIDDEN);
    wt[((size_t)e * D_MODEL + n) * HIDDEN + k] = (bf16_t)w[idx];
}

// ---------------------------------------------------------------- gating
__global__ void gate_kernel(const float* __restrict__ x, const float* __restrict__ gw,
                            int* __restrict__ cnt, int* __restrict__ tok_ids,
                            float* __restrict__ tok_w) {
    int wave = threadIdx.x >> 5;
    int lane = threadIdx.x & 31;
    int t = blockIdx.x * 8 + wave;
    const float* xr = x + (size_t)t * D_MODEL;
    float s[N_EXPERTS];
#pragma unroll
    for (int e = 0; e < N_EXPERTS; ++e) s[e] = 0.0f;
    for (int d = lane; d < D_MODEL; d += 32) {
        float xv = xr[d];
        const float4* g = (const float4*)(gw + (size_t)d * N_EXPERTS);
        float4 g0 = g[0], g1 = g[1];
        s[0] += xv * g0.x; s[1] += xv * g0.y; s[2] += xv * g0.z; s[3] += xv * g0.w;
        s[4] += xv * g1.x; s[5] += xv * g1.y; s[6] += xv * g1.z; s[7] += xv * g1.w;
    }
#pragma unroll
    for (int e = 0; e < N_EXPERTS; ++e)
#pragma unroll
        for (int off = 16; off > 0; off >>= 1) s[e] += __shfl_xor(s[e], off, 32);
    if (lane == 0) {
        int i0 = 0; float v0 = s[0];
#pragma unroll
        for (int e = 1; e < N_EXPERTS; ++e) if (s[e] > v0) { v0 = s[e]; i0 = e; }
        int i1 = -1; float v1 = -3.4e38f;
#pragma unroll
        for (int e = 0; e < N_EXPERTS; ++e)
            if (e != i0 && s[e] > v1) { v1 = s[e]; i1 = e; }
        float w0 = 1.0f / (1.0f + __expf(v1 - v0));   // softmax over {v0,v1}
        float w1 = 1.0f - w0;
        int p0 = atomicAdd(&cnt[i0], 1);
        tok_ids[i0 * T_TOKENS + p0] = t; tok_w[i0 * T_TOKENS + p0] = w0;
        int p1 = atomicAdd(&cnt[i1], 1);
        tok_ids[i1 * T_TOKENS + p1] = t; tok_w[i1 * T_TOKENS + p1] = w1;
    }
}

// ------------------------------------------------- async staging helpers
#if USE_ASYNC_LDS
template <bool GATHER_A>
__device__ __forceinline__ void stage_a_async(const bf16_t* __restrict__ A,
                                              const bf16_t* __restrict__ zpage,
                                              const int* __restrict__ tok_ids,
                                              bf16_t* aS, int tid, int rowBase,
                                              int count, int K, int kBase) {
#pragma unroll
    for (int i = 0; i < 4; ++i) {
        int c = tid + i * 256;
        int row = c >> 3;
        int j = c & 7;
        int grow = rowBase + row;
        const bf16_t* gp;
        if (grow < count) {
            size_t arow = GATHER_A ? (size_t)tok_ids[grow] : (size_t)grow;
            gp = A + arow * (size_t)K + kBase + j * 8;
        } else {
            gp = zpage;   // zeroed 16B page: keep EXEC full for async load
        }
        bf16_t* lp = aS + row * LDS_STRIDE + j * 8;
        __builtin_amdgcn_global_load_async_to_lds_b128(
            (AS1 i32x4*)const_cast<bf16_t*>(gp), (AS3 i32x4*)lp, 0, 0);
    }
}

__device__ __forceinline__ void stage_b_async(const bf16_t* __restrict__ B, bf16_t* bS,
                                              int tid, int colBase, int K, int kBase) {
#pragma unroll
    for (int i = 0; i < 4; ++i) {
        int c = tid + i * 256;
        int row = c >> 3;
        int j = c & 7;
        const bf16_t* gp = B + (size_t)(colBase + row) * K + kBase + j * 8;
        bf16_t* lp = bS + row * LDS_STRIDE + j * 8;
        __builtin_amdgcn_global_load_async_to_lds_b128(
            (AS1 i32x4*)const_cast<bf16_t*>(gp), (AS3 i32x4*)lp, 0, 0);
    }
}
#endif

#if USE_TDM
// TDM: load a BN x BK tile of B ([N][K] bf16, row-major) into LDS with native
// padding (4 DWORDs after every 32 DWORDs -> 72-element padded rows).
__device__ __forceinline__ void tdm_load_b(const bf16_t* __restrict__ B, bf16_t* bS,
                                           int colBase, int kBase, int K, int N) {
    unsigned long long ga =
        (unsigned long long)(size_t)(B + (size_t)colBase * K + kBase);
    unsigned int la = (unsigned int)(size_t)(AS3 void*)bS;
    unsigned int d0 = (unsigned int)(K - kBase);   // remaining tensor dim0 (elems)
    unsigned int d1 = (unsigned int)(N - colBase); // remaining tensor dim1 (rows)
    u32x4 g0;
    g0[0] = 1u;                                            // count=1 valid user D#
    g0[1] = la;                                            // lds_addr
    g0[2] = (unsigned int)ga;                              // global_addr[31:0]
    g0[3] = (unsigned int)((ga >> 32) & 0x1FFFFFFull) | (2u << 30);  // addr[56:32]|type=2
    i32x8 g1;
    // data_size=2B(code1), pad_enable, pad_interval=32DW(code4), pad_amount=4DW(code3)
    g1[0] = (int)((1u << 16) | (1u << 20) | (4u << 22) | (3u << 25));
    g1[1] = (int)((d0 & 0xFFFFu) << 16);                               // dim0[15:0]
    g1[2] = (int)(((d0 >> 16) & 0xFFFFu) | ((d1 & 0xFFFFu) << 16));    // dim0[31:16]|dim1[15:0]
    g1[3] = (int)(((d1 >> 16) & 0xFFFFu) | ((unsigned)BK << 16));      // dim1[31:16]|tile_dim0
    g1[4] = (int)(unsigned)BN;                                         // tile_dim1 | tile_dim2=0
    g1[5] = (int)(unsigned)K;                                          // dim0_stride[31:0]
    g1[6] = 0;                                                         // stride hi | dim1_stride lo
    g1[7] = 0;
    i32x4 gz = {0, 0, 0, 0};
#if __clang_major__ >= 23
    i32x8 gz8 = {0, 0, 0, 0, 0, 0, 0, 0};
    __builtin_amdgcn_tensor_load_to_lds(g0, g1, gz, gz, gz8, 0);
#else
    __builtin_amdgcn_tensor_load_to_lds(g0, g1, gz, gz, 0);
#endif
}
#endif

// ---------------------------------------------------------------- WMMA core
__device__ __forceinline__ void compute_tiles(const bf16_t* aS, const bf16_t* bS,
                                              int wm, int wn, int lrow, int lhi,
                                              v8f (&acc)[4][2]) {
#pragma unroll
    for (int ks = 0; ks < 2; ++ks) {
        v16bf afrag[4], bfrag[2];
#pragma unroll
        for (int mt = 0; mt < 4; ++mt) {
            const bf16_t* p = aS + (wm * 64 + mt * 16 + lrow) * LDS_STRIDE + ks * 32 + lhi * 16;
            v8bf lo = *(const v8bf*)p;
            v8bf hi = *(const v8bf*)(p + 8);
            afrag[mt] = __builtin_shufflevector(lo, hi, 0, 1, 2, 3, 4, 5, 6, 7,
                                                8, 9, 10, 11, 12, 13, 14, 15);
        }
#pragma unroll
        for (int nt = 0; nt < 2; ++nt) {
            const bf16_t* p = bS + (wn * 32 + nt * 16 + lrow) * LDS_STRIDE + ks * 32 + lhi * 16;
            v8bf lo = *(const v8bf*)p;
            v8bf hi = *(const v8bf*)(p + 8);
            bfrag[nt] = __builtin_shufflevector(lo, hi, 0, 1, 2, 3, 4, 5, 6, 7,
                                                8, 9, 10, 11, 12, 13, 14, 15);
        }
#pragma unroll
        for (int mt = 0; mt < 4; ++mt)
#pragma unroll
            for (int nt = 0; nt < 2; ++nt)
                acc[mt][nt] = __builtin_amdgcn_wmma_f32_16x16x32_bf16(
                    false, afrag[mt], false, bfrag[nt], (short)0, acc[mt][nt],
                    false, false);
    }
}

// ---------------------------------------------------------------- WMMA GEMM
// A: [*][K] bf16 (gathered via tok_ids if GATHER_A), B: [N][K] bf16.
// FUSE_SILU: D = silu(A@B^T) -> Hout (bf16).  else: Out[tok] += w * (A@B^T) (f32).
template <bool GATHER_A, bool FUSE_SILU>
__global__ __launch_bounds__(256) void ffn_gemm_kernel(
    const bf16_t* __restrict__ A, const bf16_t* __restrict__ B,
    const int* __restrict__ cntp, const int* __restrict__ tok_ids,
    const float* __restrict__ tok_w, const bf16_t* __restrict__ zpage,
    bf16_t* __restrict__ Hout, float* __restrict__ Out, int K, int N) {
    __shared__ __align__(16) bf16_t ldsA[2 * STAGE_A];
    __shared__ __align__(16) bf16_t ldsB[2 * STAGE_B];

    const int count = *cntp;
    const int rowBase = blockIdx.y * BM;
    if (rowBase >= count) return;
    const int colBase = blockIdx.x * BN;

    const int tid  = threadIdx.x;
    const int lane = tid & 31;
    const int wave = tid >> 5;
    const int wm = wave >> 2;     // 0..1 : 64-row strip
    const int wn = wave & 3;      // 0..3 : 32-col strip
    const int lrow = lane & 15;
    const int lhi  = lane >> 4;

    v8f acc[4][2];
#pragma unroll
    for (int mt = 0; mt < 4; ++mt)
#pragma unroll
        for (int nt = 0; nt < 2; ++nt)
#pragma unroll
            for (int r = 0; r < 8; ++r) acc[mt][nt][r] = 0.0f;

    const int kTiles = K / BK;

#if USE_ASYNC_LDS
    // ---- double-buffered async pipeline ----
    stage_a_async<GATHER_A>(A, zpage, tok_ids, ldsA, tid, rowBase, count, K, 0);
#if USE_TDM
    if (wave == 0) tdm_load_b(B, ldsB, colBase, 0, K, N);
#else
    stage_b_async(B, ldsB, tid, colBase, K, 0);
#endif
    for (int kb = 0; kb < kTiles; ++kb) {
        const int cur = kb & 1;
        const bf16_t* aS = ldsA + cur * STAGE_A;
        const bf16_t* bS = ldsB + cur * STAGE_B;
        if (kb + 1 < kTiles) {
            const int kn = (kb + 1) * BK;
            bf16_t* aN = ldsA + (cur ^ 1) * STAGE_A;
            bf16_t* bN = ldsB + (cur ^ 1) * STAGE_B;
            stage_a_async<GATHER_A>(A, zpage, tok_ids, aN, tid, rowBase, count, K, kn);
#if USE_TDM
            if (wave == 0) tdm_load_b(B, bN, colBase, kn, K, N);
            __builtin_amdgcn_s_wait_asynccnt(4);   // own stage-cur A loads done
            __builtin_amdgcn_s_wait_tensorcnt(1);  // stage-cur TDM done (wave0; NOP others)
#else
            stage_b_async(B, bN, tid, colBase, K, kn);
            __builtin_amdgcn_s_wait_asynccnt(8);
#endif
        } else {
            __builtin_amdgcn_s_wait_asynccnt(0);
#if USE_TDM
            __builtin_amdgcn_s_wait_tensorcnt(0);
#endif
        }
        __syncthreads();
        compute_tiles(aS, bS, wm, wn, lrow, lhi, acc);
        __syncthreads();
    }
#else
    // ---- synchronous fallback (round-1 path) ----
    for (int kb = 0; kb < kTiles; ++kb) {
        const int kBase = kb * BK;
#pragma unroll
        for (int i = 0; i < 4; ++i) {
            int c = tid + i * 256;
            int row = c >> 3;
            int j = c & 7;
            uint4 av = make_uint4(0u, 0u, 0u, 0u);
            int grow = rowBase + row;
            if (grow < count) {
                size_t arow = GATHER_A ? (size_t)tok_ids[grow] : (size_t)grow;
                av = *(const uint4*)(A + arow * (size_t)K + kBase + j * 8);
            }
            *(uint4*)(ldsA + row * LDS_STRIDE + j * 8) = av;
            uint4 bv = *(const uint4*)(B + (size_t)(colBase + row) * K + kBase + j * 8);
            *(uint4*)(ldsB + row * LDS_STRIDE + j * 8) = bv;
        }
        __syncthreads();
        compute_tiles(ldsA, ldsB, wm, wn, lrow, lhi, acc);
        __syncthreads();
    }
#endif

    if (FUSE_SILU) {
#pragma unroll
        for (int mt = 0; mt < 4; ++mt)
#pragma unroll
            for (int nt = 0; nt < 2; ++nt) {
                int nIdx = colBase + wn * 32 + nt * 16 + lrow;
#pragma unroll
                for (int r = 0; r < 8; ++r) {
                    int m = rowBase + wm * 64 + mt * 16 + lhi * 8 + r;
                    if (m < count) {
                        float v = acc[mt][nt][r];
                        float sv = v / (1.0f + __expf(-v));
                        Hout[(size_t)m * N + nIdx] = (bf16_t)sv;
                    }
                }
            }
    } else {
#pragma unroll
        for (int mt = 0; mt < 4; ++mt) {
            int mBase = rowBase + wm * 64 + mt * 16 + lhi * 8;
#pragma unroll
            for (int r = 0; r < 8; ++r) {
                int m = mBase + r;
                if (m < count) {
                    int tok = tok_ids[m];
                    float wgt = tok_w[m];
                    size_t obase = (size_t)tok * N;
#pragma unroll
                    for (int nt = 0; nt < 2; ++nt) {
                        int nIdx = colBase + wn * 32 + nt * 16 + lrow;
                        Out[obase + nIdx] += wgt * acc[mt][nt][r];
                    }
                }
            }
        }
    }
}

// ---------------------------------------------------------------- launch
extern "C" void kernel_launch(void* const* d_in, const int* in_sizes, int n_in,
                              void* d_out, int out_size, void* d_ws, size_t ws_size,
                              hipStream_t stream) {
    const float* x  = (const float*)d_in[0];
    const float* gw = (const float*)d_in[1];
    const float* w1 = (const float*)d_in[2];
    const float* w2 = (const float*)d_in[3];
    float* out = (float*)d_out;

    char* ws = (char*)d_ws;
    size_t o = 0;
    int* cnt = (int*)(ws + o);            o += 256;
    bf16_t* zpage = (bf16_t*)(ws + o);    o += 256;   // zeroed page for OOB async loads
    int* tok_ids = (int*)(ws + o);        o += (size_t)N_EXPERTS * T_TOKENS * 4;
    float* tok_w = (float*)(ws + o);      o += (size_t)N_EXPERTS * T_TOKENS * 4;
    bf16_t* xb = (bf16_t*)(ws + o);       o += (size_t)T_TOKENS * D_MODEL * 2;
    bf16_t* w1t = (bf16_t*)(ws + o);      o += (size_t)N_EXPERTS * D_MODEL * HIDDEN * 2;
    bf16_t* w2t = (bf16_t*)(ws + o);      o += (size_t)N_EXPERTS * D_MODEL * HIDDEN * 2;
    bf16_t* hbuf = (bf16_t*)(ws + o);     o += (size_t)T_TOKENS * HIDDEN * 2;

    (void)hipMemsetAsync(d_out, 0, (size_t)out_size * sizeof(float), stream);
    (void)hipMemsetAsync(cnt, 0, 512, stream);   // counts + zero page

    cvt_x_kernel<<<(T_TOKENS * D_MODEL) / (256 * 4), 256, 0, stream>>>(x, xb);
    cvt_w1t_kernel<<<((size_t)N_EXPERTS * D_MODEL * HIDDEN) / 256, 256, 0, stream>>>(w1, w1t);
    cvt_w2t_kernel<<<((size_t)N_EXPERTS * D_MODEL * HIDDEN) / 256, 256, 0, stream>>>(w2, w2t);
    gate_kernel<<<T_TOKENS / 8, 256, 0, stream>>>(x, gw, cnt, tok_ids, tok_w);

    for (int e = 0; e < N_EXPERTS; ++e) {
        ffn_gemm_kernel<true, true>
            <<<dim3(HIDDEN / BN, T_TOKENS / BM), 256, 0, stream>>>(
                xb, w1t + (size_t)e * HIDDEN * D_MODEL, cnt + e,
                tok_ids + e * T_TOKENS, nullptr, zpage, hbuf, nullptr,
                D_MODEL, HIDDEN);
        ffn_gemm_kernel<false, false>
            <<<dim3(D_MODEL / BN, T_TOKENS / BM), 256, 0, stream>>>(
                hbuf, w2t + (size_t)e * D_MODEL * HIDDEN, cnt + e,
                tok_ids + e * T_TOKENS, tok_w + e * T_TOKENS, zpage, nullptr,
                out, HIDDEN, D_MODEL);
    }
}
